// QuantumHDCEncoder_54047868452943
// MI455X (gfx1250) — matrix operand order repaired and checked by reading, not compiled, not run on hardware
//
#include <hip/hip_runtime.h>

// Problem dims (from reference)
#define BATCH   1024
#define KDIM    4096
#define NDIM    8192
#define NANG    39      // QUANTUM_DEPTH * NUM_QUBITS

// GEMM tiling
#define BM 128
#define BN 128
#define BK 32
#define NKB (KDIM / BK)     // 128 k-chunks
#define LDSS 40             // padded bf16 row stride (80 B, 16B-aligned, conflict-free)

typedef __attribute__((ext_vector_type(16))) __bf16 v16bf;
typedef __attribute__((ext_vector_type(8)))  float  v8f;

union FragBf { uint4 q[2]; v16bf v; };

// Low 32 bits of a generic LDS pointer == LDS byte offset
__device__ __forceinline__ uint32_t lds_off(const void* p) {
  return (uint32_t)(uintptr_t)p;
}

// Per-lane async DMA: 16 bytes global -> LDS, tracked by ASYNCcnt.
__device__ __forceinline__ void async_copy16(uint32_t ldsOff, const void* g) {
  asm volatile("global_load_async_to_lds_b128 %0, %1, off"
               :: "v"(ldsOff), "v"((uint64_t)(uintptr_t)g)
               : "memory");
}
__device__ __forceinline__ void wait_async0() {
  asm volatile("s_wait_asynccnt 0x0" ::: "memory");
}

// Stage one 128-row x 64B bf16 plane into padded LDS (stride 80B).
// 512 chunk-tasks of 16B; 256 threads -> 2 async issues per thread.
__device__ __forceinline__ void stage_plane(const __bf16* __restrict__ g,  // at [row0][kb]
                                            uint32_t ldsBase, int t) {
#pragma unroll
  for (int i = 0; i < 2; ++i) {
    const int task = t + i * 256;
    const int row = task >> 2;
    const int ch  = task & 3;
    const void* ga = (const char*)g + (size_t)row * (KDIM * 2) + ch * 16;
    async_copy16(ldsBase + row * 80 + ch * 16, ga);
  }
}

// ---------------------------------------------------------------------------
// Kernel 0: one-time f32 -> bf16 (hi, lo) split.  x and W are split once and
// reused by all GEMM tiles (x by 64 N-blocks, W by 8 M-blocks).
// ---------------------------------------------------------------------------
__global__ __launch_bounds__(256) void split_bf16_kernel(
    const float* __restrict__ src, __bf16* __restrict__ hi,
    __bf16* __restrict__ lo, int n8) {
  const int i = blockIdx.x * blockDim.x + threadIdx.x;
  if (i >= n8) return;
  alignas(16) float f[8];
  *(float4*)&f[0] = *((const float4*)src + (size_t)i * 2 + 0);
  *(float4*)&f[4] = *((const float4*)src + (size_t)i * 2 + 1);
  alignas(16) __bf16 h[8], l[8];
#pragma unroll
  for (int j = 0; j < 8; ++j) {
    __bf16 a = (__bf16)f[j];
    h[j] = a;
    l[j] = (__bf16)(f[j] - (float)a);
  }
  *(uint4*)(hi + (size_t)i * 8) = *(uint4*)h;
  *(uint4*)(lo + (size_t)i * 8) = *(uint4*)l;
}

// ---------------------------------------------------------------------------
// Kernel 1: C = x @ W^T, 3xBF16 split-precision WMMA, async double-buffered
// LDS staging.  Block: 256 threads = 8 waves (2x4); tile 128x128; each wave
// owns 64x32 = 4x2 WMMA tiles -> 24 WMMAs per k-chunk per wave.
// ---------------------------------------------------------------------------
__global__ __launch_bounds__(256) void gemm_bf16x3_async_kernel(
    const __bf16* __restrict__ Xhi, const __bf16* __restrict__ Xlo,  // [BATCH, KDIM]
    const __bf16* __restrict__ Whi, const __bf16* __restrict__ Wlo,  // [NDIM,  KDIM]
    float* __restrict__ C)                                            // [BATCH, NDIM]
{
  __shared__ __bf16 sA[2][2][BM * LDSS];   // [buf][hi/lo]
  __shared__ __bf16 sB[2][2][BN * LDSS];

  const int t    = threadIdx.x;
  const int lane = t & 31;
  const int wave = t >> 5;
  const int wm   = wave >> 2;   // 0..1 (M direction, 64 rows each)
  const int wn   = wave & 3;    // 0..3 (N direction, 32 cols each)

  const int m0blk = blockIdx.y * BM;
  const int n0blk = blockIdx.x * BN;

  const int khalf = lane >> 4;  // which K-half this lane holds
  const int lrow  = lane & 15;  // M row (A) / N col (B) within a 16-tile

  const __bf16* xh = Xhi + (size_t)m0blk * KDIM;
  const __bf16* xl = Xlo + (size_t)m0blk * KDIM;
  const __bf16* wh = Whi + (size_t)n0blk * KDIM;
  const __bf16* wl = Wlo + (size_t)n0blk * KDIM;

  v8f acc[4][2] = {};

  auto issue = [&](int kb, int buf) {
    stage_plane(xh + kb, lds_off(&sA[buf][0][0]), t);
    stage_plane(xl + kb, lds_off(&sA[buf][1][0]), t);
    stage_plane(wh + kb, lds_off(&sB[buf][0][0]), t);
    stage_plane(wl + kb, lds_off(&sB[buf][1][0]), t);
  };

  issue(0, 0);  // prologue

  for (int kbi = 0; kbi < NKB; ++kbi) {
    const int buf = kbi & 1;

    wait_async0();      // my copies for buf are done (in-order completion)
    __syncthreads();    // everyone's copies done + everyone finished prev compute

    if (kbi + 1 < NKB)  // DMA for next chunk overlaps this chunk's WMMAs
      issue((kbi + 1) * BK, buf ^ 1);

    // B fragments (bf16 32x16: lane = N col, K = khalf*16 .. +15 contiguous)
    FragBf bfr[2][2];   // [plane][nt]
#pragma unroll
    for (int pl = 0; pl < 2; ++pl)
#pragma unroll
      for (int nt = 0; nt < 2; ++nt) {
        const __bf16* p = &sB[buf][pl][(wn * 32 + nt * 16 + lrow) * LDSS + khalf * 16];
        bfr[pl][nt].q[0] = *(const uint4*)(p);
        bfr[pl][nt].q[1] = *(const uint4*)(p + 8);
      }

#pragma unroll
    for (int mt = 0; mt < 4; ++mt) {
      // A fragments (bf16 16x32: lane = M row, K = khalf*16 .. +15 contiguous)
      const int abase = (wm * 64 + mt * 16 + lrow) * LDSS + khalf * 16;
      FragBf ah, al;
      ah.q[0] = *(const uint4*)&sA[buf][0][abase];
      ah.q[1] = *(const uint4*)&sA[buf][0][abase + 8];
      al.q[0] = *(const uint4*)&sA[buf][1][abase];
      al.q[1] = *(const uint4*)&sA[buf][1][abase + 8];
#pragma unroll
      for (int nt = 0; nt < 2; ++nt) {
        acc[mt][nt] = __builtin_amdgcn_wmma_f32_16x16x32_bf16(
            false, ah.v, false, bfr[0][nt].v, (short)0, acc[mt][nt], false, false);
        acc[mt][nt] = __builtin_amdgcn_wmma_f32_16x16x32_bf16(
            false, ah.v, false, bfr[1][nt].v, (short)0, acc[mt][nt], false, false);
        acc[mt][nt] = __builtin_amdgcn_wmma_f32_16x16x32_bf16(
            false, al.v, false, bfr[0][nt].v, (short)0, acc[mt][nt], false, false);
      }
    }
  }

  // Epilogue: C/D layout (VGPR r, lane): row = r + 8*(lane>=16), col = lane&15
#pragma unroll
  for (int mt = 0; mt < 4; ++mt)
#pragma unroll
    for (int nt = 0; nt < 2; ++nt) {
      const int grow = m0blk + wm * 64 + mt * 16 + khalf * 8;
      const int gcol = n0blk + wn * 32 + nt * 16 + lrow;
#pragma unroll
      for (int r = 0; r < 8; ++r)
        C[(size_t)(grow + r) * NDIM + gcol] = acc[mt][nt][r];
    }
}

// ---------------------------------------------------------------------------
// Kernel 2: fused row-normalization + 39 sequential roll-mix gates + complex
// output.  One block per batch row; the full 8192-element row stays in
// registers (8 complex per thread across 1024 threads); only the per-thread
// boundary element goes through LDS each step.
// new[i] = cos(a/2)*amp[i] + i*sin(a/2)*amp[i-1]  (cyclic roll by 1)
// ---------------------------------------------------------------------------
__global__ __launch_bounds__(1024) void norm_gates_kernel(
    const float* __restrict__ C,    // [BATCH, NDIM]
    const float* __restrict__ qp,   // [NANG] angles, depth-major
    float* __restrict__ out)        // [BATCH, NDIM] complex64 (interleaved)
{
  __shared__ float  red[32];
  __shared__ float2 edge[1024];

  const int row = blockIdx.x;
  const int t   = threadIdx.x;

  const float* cp = C + (size_t)row * NDIM + t * 8;
  float re[8], im[8];
  float ss = 0.f;
#pragma unroll
  for (int j = 0; j < 8; ++j) {
    float v = cp[j];
    re[j] = v; im[j] = 0.f;
    ss += v * v;
  }
#pragma unroll
  for (int off = 16; off > 0; off >>= 1) ss += __shfl_down(ss, off, 32);
  if ((t & 31) == 0) red[t >> 5] = ss;
  __syncthreads();
  if (t < 32) {
    float v = red[t];
#pragma unroll
    for (int off = 16; off > 0; off >>= 1) v += __shfl_down(v, off, 32);
    if (t == 0) red[0] = v;
  }
  __syncthreads();
  const float inv = rsqrtf(red[0]);
#pragma unroll
  for (int j = 0; j < 8; ++j) re[j] *= inv;   // im == 0; second norm is identity

  for (int d = 0; d < NANG; ++d) {
    const float ang = 0.5f * qp[d];
    const float c = cosf(ang);
    const float s = sinf(ang);

    edge[t] = make_float2(re[7], im[7]);        // my last (old) element
    __syncthreads();
    const float2 pv = edge[(t + 1023) & 1023];  // previous thread's last, cyclic
    __syncthreads();

#pragma unroll
    for (int j = 7; j >= 1; --j) {              // descending: a[j-1] still old
      const float nr = c * re[j] - s * im[j - 1];
      const float ni = c * im[j] + s * re[j - 1];
      re[j] = nr; im[j] = ni;
    }
    const float nr0 = c * re[0] - s * pv.y;
    const float ni0 = c * im[0] + s * pv.x;
    re[0] = nr0; im[0] = ni0;
  }

  float2* op = (float2*)out + (size_t)row * NDIM + t * 8;
#pragma unroll
  for (int j = 0; j < 8; ++j) op[j] = make_float2(re[j], im[j]);
}

// ---------------------------------------------------------------------------
// Workspace layout (requires ws_size >= 176 MB):
//   C    : 1024*8192 f32   = 32 MB
//   Xhi  : 1024*4096 bf16  =  8 MB
//   Xlo  : 1024*4096 bf16  =  8 MB
//   Whi  : 8192*4096 bf16  = 64 MB
//   Wlo  : 8192*4096 bf16  = 64 MB
// ---------------------------------------------------------------------------
extern "C" void kernel_launch(void* const* d_in, const int* in_sizes, int n_in,
                              void* d_out, int out_size, void* d_ws, size_t ws_size,
                              hipStream_t stream) {
  const float* x  = (const float*)d_in[0];   // [1024, 4096]
  const float* W  = (const float*)d_in[1];   // [8192, 4096]
  const float* qp = (const float*)d_in[2];   // [39]

  char* ws = (char*)d_ws;
  float*  C   = (float*)ws;                      ws += (size_t)BATCH * NDIM * 4;
  __bf16* Xhi = (__bf16*)ws;                     ws += (size_t)BATCH * KDIM * 2;
  __bf16* Xlo = (__bf16*)ws;                     ws += (size_t)BATCH * KDIM * 2;
  __bf16* Whi = (__bf16*)ws;                     ws += (size_t)NDIM  * KDIM * 2;
  __bf16* Wlo = (__bf16*)ws;
  float* out = (float*)d_out;                    // complex64 interleaved

  const int nx8 = (BATCH * KDIM) / 8;
  const int nw8 = (NDIM * KDIM) / 8;
  split_bf16_kernel<<<(nx8 + 255) / 256, 256, 0, stream>>>(x, Xhi, Xlo, nx8);
  split_bf16_kernel<<<(nw8 + 255) / 256, 256, 0, stream>>>(W, Whi, Wlo, nw8);

  dim3 ggrid(NDIM / BN, BATCH / BM);             // (64, 8)
  gemm_bf16x3_async_kernel<<<ggrid, 256, 0, stream>>>(Xhi, Xlo, Whi, Wlo, C);
  norm_gates_kernel<<<BATCH, 1024, 0, stream>>>(C, qp, out);
}
